// TrackQueryBased_13005160972700
// MI455X (gfx1250) — matrix operand order repaired and checked by reading, not compiled, not run on hardware
//
#include <hip/hip_runtime.h>

// ---------------------------------------------------------------------------
// Problem constants
// ---------------------------------------------------------------------------
#define NPTS   100000
#define CDIM   256
#define HEADS  4
#define DH     64
#define TOPKN  256
#define KBLK   128
#define NBLKS  782              // ceil(NPTS/128)
#define NPAD   (NBLKS * KBLK)   // 100096
#define CHUNKS 28               // key chunks per (h, qtile)
#define FSTR   264              // LDS row stride (u16) feat tile   (16 rows -> distinct banks)
#define WSTR   264              // LDS row stride (u16) weight tile
#define KSTR   72               // LDS row stride (u16) K tile [128 keys][64 d]
#define VTSTR  136              // LDS row stride (u16) V^T tile [64 d][128 keys]
#define PSTR   136              // LDS row stride (u16) P tile [16 q][128 keys]

typedef __attribute__((ext_vector_type(16))) __bf16 v16bf;
typedef __attribute__((ext_vector_type(8)))  float  v8f;

union FragB { unsigned u[8]; uint4 q[2]; v16bf v; };

__device__ inline unsigned short f2bf(float f) {
  unsigned x = __builtin_bit_cast(unsigned, f);
  unsigned r = (x + 0x7FFFu + ((x >> 16) & 1u)) >> 16;   // RNE
  return (unsigned short)r;
}

__device__ inline v8f wmma_bf16(const FragB& a, const FragB& b, v8f c) {
  return __builtin_amdgcn_wmma_f32_16x16x32_bf16(false, a.v, false, b.v,
                                                 (short)0, c, false, false);
}

// --- CDNA5 async global->LDS copy (ASYNCcnt-tracked, bypasses VGPRs) -------
// ISA 15.18.3: GLOBAL_LOAD_ASYNC_TO_LDS_B128, GV mode (64-bit vaddr, saddr=off)
// LDS address is taken from the vdst VGPR; generic shared-pointer low 32 bits = LDS off.
__device__ inline void async_g2l_b128(const unsigned short* lds_dst,
                                      const unsigned short* gsrc) {
  unsigned ldsOff = (unsigned)(size_t)lds_dst;
  unsigned long long ga = (unsigned long long)(size_t)gsrc;
  asm volatile("global_load_async_to_lds_b128 %0, %1, off"
               :: "v"(ldsOff), "v"(ga) : "memory");
}
__device__ inline void wait_async0() {
  asm volatile("s_wait_asynccnt 0x0" ::: "memory");
}

// A-frag (16x32, 16-bit): lane m=lane&15, hf=lane>>4 needs
//   k in [8*hf, 8*hf+8) U [16+8*hf, 16+8*hf+8)  -> two 16B runs.
// B-frag (32x16, 16-bit): lane n=lane&15, hf=lane>>4 needs
//   k in [8*hf, 8*hf+8) U [16+8*hf, 16+8*hf+8)  -> two 16B runs ([n][k] storage).
__device__ inline void loadA(FragB& f, const unsigned short* p, int kbase, int hf) {
  f.q[0] = *(const uint4*)(p + kbase + 8 * hf);
  f.q[1] = *(const uint4*)(p + kbase + 16 + 8 * hf);
}
__device__ inline void loadB(FragB& f, const unsigned short* p, int kbase, int hf) {
  f.q[0] = *(const uint4*)(p + kbase + 8 * hf);
  f.q[1] = *(const uint4*)(p + kbase + 16 + 8 * hf);
}

__device__ inline unsigned mkkey(float f) {
  unsigned u = __builtin_bit_cast(unsigned, f);
  return (u & 0x80000000u) ? ~u : (u | 0x80000000u);
}

// ---------------------------------------------------------------------------
// fp32 [K][N] -> bf16 transposed [N][K]  (weights become B-operand friendly)
// ---------------------------------------------------------------------------
__global__ void k_cvt_bf16_t(const float* __restrict__ s,
                             unsigned short* __restrict__ d) {
  int t = blockIdx.x * blockDim.x + threadIdx.x;    // 65536
  if (t < CDIM * CDIM) {
    int n = t >> 8, k = t & 255;
    d[n * CDIM + k] = f2bf(s[k * CDIM + n]);
  }
}

// plain fp32 -> bf16 (same layout)
__global__ void k_cvt_bf16(const float* __restrict__ s,
                           unsigned short* __restrict__ d, int n) {
  int i = blockIdx.x * blockDim.x + threadIdx.x;
  int st = gridDim.x * blockDim.x;
  for (; i < n; i += st) d[i] = f2bf(s[i]);
}

// ---------------------------------------------------------------------------
// Fused objectness + K/V projection over all N points.
// grid = NBLKS (128 rows each), block = 256 (8 waves, one 16-row tile each).
// Dynamic LDS: feat tile [128][FSTR] bf16 + one transposed weight [256][WSTR].
// Outputs: obj[N], K [H][NPAD][D] bf16, V^T [H][D][NPAD] bf16.
// ---------------------------------------------------------------------------
__global__ __launch_bounds__(256) void k_fused_nkv(
    const float* __restrict__ feat,
    const unsigned short* __restrict__ w1t, const unsigned short* __restrict__ wkt,
    const unsigned short* __restrict__ wvt,
    const float* __restrict__ b1, const float* __restrict__ bkk,
    const float* __restrict__ bvv,
    const float* __restrict__ w2, const float* __restrict__ b2,
    float* __restrict__ obj, unsigned short* __restrict__ Kn,
    unsigned short* __restrict__ Vt) {
  extern __shared__ unsigned short sm[];
  unsigned short* sf = sm;                   // [128][FSTR]
  unsigned short* sw = sm + 128 * FSTR;      // [256][WSTR]  (transposed weight)
  const int tid = threadIdx.x;
  const int rowBase = blockIdx.x * 128;

  { // stage feat block as bf16 (needs conversion -> regular path)
    int r = tid >> 1, ch = (tid & 1) * 128;
    int grow = rowBase + r;
    for (int c = 0; c < 128; c += 4) {
      float4 v = {0.f, 0.f, 0.f, 0.f};
      if (grow < NPTS)
        v = *(const float4*)(feat + (size_t)grow * CDIM + ch + c);
      sf[r * FSTR + ch + c + 0] = f2bf(v.x);
      sf[r * FSTR + ch + c + 1] = f2bf(v.y);
      sf[r * FSTR + ch + c + 2] = f2bf(v.z);
      sf[r * FSTR + ch + c + 3] = f2bf(v.w);
    }
  }
  __syncthreads();

  const int lane = tid & 31, wvid = tid >> 5, hf = lane >> 4, ml = lane & 15;
  FragB af[8];
#pragma unroll
  for (int kb = 0; kb < 8; ++kb)
    loadA(af[kb], sf + (wvid * 16 + ml) * FSTR, kb * 32, hf);

  float oacc[8];
#pragma unroll
  for (int i = 0; i < 8; ++i) oacc[i] = 0.f;

  for (int w = 0; w < 3; ++w) {
    const unsigned short* W = (w == 0) ? w1t : (w == 1) ? wkt : wvt;
    __syncthreads();
    // stage transposed weight into LDS via CDNA5 async global->LDS DMA
#pragma unroll
    for (int q = 0; q < 32; ++q)
      async_g2l_b128(sw + tid * WSTR + q * 8, W + tid * CDIM + q * 8);
    wait_async0();
    __syncthreads();

    const float* bias = (w == 0) ? b1 : (w == 1) ? bkk : bvv;
    float bvals[16], wvals[16];
#pragma unroll
    for (int nt = 0; nt < 16; ++nt) bvals[nt] = bias[nt * 16 + ml];
    if (w == 0) {
#pragma unroll
      for (int nt = 0; nt < 16; ++nt) wvals[nt] = w2[nt * 16 + ml];
    }

    for (int nt = 0; nt < 16; ++nt) {
      int ncol = nt * 16 + ml;
      v8f acc = {};
      for (int kb = 0; kb < 8; ++kb) {
        FragB bf;
        loadB(bf, sw + ncol * WSTR, kb * 32, hf);
        acc = wmma_bf16(af[kb], bf, acc);
      }
      float bvs = bvals[nt];
      if (w == 0) {
        float wc = wvals[nt];
#pragma unroll
        for (int i = 0; i < 8; ++i) {
          float hv = acc[i] + bvs;
          hv = hv > 0.f ? hv : 0.f;               // relu
          oacc[i] += hv * wc;                     // dot with obj_w2
        }
      } else {
        int hh = ncol >> 6, dd = ncol & 63;
#pragma unroll
        for (int i = 0; i < 8; ++i) {
          int grow = rowBase + wvid * 16 + i + 8 * hf;
          if (grow < NPTS) {
            unsigned short bb = f2bf(acc[i] + bvs);
            if (w == 1) Kn[((size_t)hh * NPAD + grow) * DH + dd] = bb;   // [H][N][D]
            else        Vt[(size_t)(hh * DH + dd) * NPAD + grow] = bb;   // [H][D][N]
          }
        }
      }
    }
    if (w == 0) { // finish objectness: reduce across 16 lanes (columns)
#pragma unroll
      for (int i = 0; i < 8; ++i) {
        float v = oacc[i];
        v += __shfl_xor(v, 1, 32);
        v += __shfl_xor(v, 2, 32);
        v += __shfl_xor(v, 4, 32);
        v += __shfl_xor(v, 8, 32);
        if (ml == i) {
          int grow = rowBase + wvid * 16 + i + 8 * hf;
          if (grow < NPTS) obj[grow] = v + b2[0];
        }
      }
    }
  }
}

// ---------------------------------------------------------------------------
// Exact top-256: 4-pass radix select
// state: st[0]=prefix, st[1]=rank, st[2]=cntA, st[3]=cntB
// ---------------------------------------------------------------------------
__global__ void k_radix_init(unsigned* st, unsigned* hist) {
  int t = threadIdx.x;
  for (int i = t; i < 1024; i += 256) hist[i] = 0;
  if (t == 0) { st[0] = 0; st[1] = TOPKN; st[2] = 0; st[3] = 0; }
}

__global__ void k_radix_hist(const float* __restrict__ obj, const unsigned* st,
                             unsigned* hist, int pass) {
  int shift = 24 - 8 * pass;
  unsigned pmask = (pass == 0) ? 0u : (0xFFFFFFFFu << (shift + 8));
  unsigned prefix = st[0];
  int i = blockIdx.x * blockDim.x + threadIdx.x;
  int s = gridDim.x * blockDim.x;
  for (; i < NPTS; i += s) {
    unsigned key = mkkey(obj[i]);
    if ((key & pmask) == (prefix & pmask))
      atomicAdd(&hist[pass * 256 + ((key >> shift) & 255u)], 1u);
  }
}

__global__ void k_radix_scan(unsigned* st, const unsigned* hist, int pass) {
  if (threadIdx.x == 0 && blockIdx.x == 0) {
    int shift = 24 - 8 * pass;
    unsigned rank = st[1], cum = 0;
    int dsel = 0;
    for (int d = 255; d >= 0; --d) {
      unsigned c = hist[pass * 256 + d];
      if (cum + c >= rank) { dsel = d; rank -= cum; break; }
      cum += c;
    }
    st[0] |= ((unsigned)dsel) << shift;
    st[1] = rank;
  }
}

__global__ void k_compact(const float* __restrict__ obj, unsigned* st,
                          unsigned* Akey, int* Aidx, int* Bidx) {
  unsigned prefix = st[0];
  int i = blockIdx.x * blockDim.x + threadIdx.x;
  int s = gridDim.x * blockDim.x;
  for (; i < NPTS; i += s) {
    unsigned key = mkkey(obj[i]);
    if (key > prefix) {
      unsigned p = atomicAdd(&st[2], 1u);
      if (p < TOPKN) { Akey[p] = key; Aidx[p] = i; }
    } else if (key == prefix) {
      unsigned p = atomicAdd(&st[3], 1u);
      if (p < 512) Bidx[p] = i;
    }
  }
}

__global__ __launch_bounds__(256) void k_finalize(
    const unsigned* st, const unsigned* Akey, const int* Aidx,
    const int* Bidx, int* idxs) {
  __shared__ unsigned sk[256];
  __shared__ int si[256];
  __shared__ int sb[512];
  int t = threadIdx.x;
  unsigned prefix = st[0];
  int r = (int)st[1];
  int nA = (int)st[2]; if (nA > TOPKN) nA = TOPKN;
  int nB = (int)st[3]; if (nB > 512)   nB = 512;
  for (int x = t; x < 512; x += 256) sb[x] = (x < nB) ? Bidx[x] : 0x7FFFFFFF;
  __syncthreads();
  for (int k = 2; k <= 512; k <<= 1)
    for (int j = k >> 1; j > 0; j >>= 1) {
      for (int x = t; x < 512; x += 256) {
        int xj = x ^ j;
        if (xj > x) {
          bool up = ((x & k) == 0);
          int a = sb[x], b = sb[xj];
          if (up ? (a > b) : (a < b)) { sb[x] = b; sb[xj] = a; }
        }
      }
      __syncthreads();
    }
  unsigned mk; int mi;
  if (t < nA)          { mk = Akey[t]; mi = Aidx[t]; }
  else if (t < nA + r) { mk = prefix;  mi = sb[t - nA]; }
  else                 { mk = 0u;      mi = 0x7FFFFFFF; }
  sk[t] = mk; si[t] = mi;
  __syncthreads();
  for (int k = 2; k <= 256; k <<= 1)
    for (int j = k >> 1; j > 0; j >>= 1) {
      int xj = t ^ j;
      if (xj > t) {
        bool up = ((t & k) == 0);
        unsigned ka = sk[t], kb2 = sk[xj];
        int ia = si[t], ib = si[xj];
        bool aBefore = (ka > kb2) || (ka == kb2 && ia < ib);
        if (up ? !aBefore : aBefore) {
          sk[t] = kb2; si[t] = ib; sk[xj] = ka; si[xj] = ia;
        }
      }
      __syncthreads();
    }
  idxs[t] = si[t];
}

// ---------------------------------------------------------------------------
// q_in = feat[idx] + (coor[idx]*0.4) @ pos_w + pos_b  (bf16 out)
// ---------------------------------------------------------------------------
__global__ void k_build_qin(const int* __restrict__ coor,
                            const float* __restrict__ feat,
                            const float* __restrict__ pos_w,
                            const float* __restrict__ pos_b,
                            const int* __restrict__ idxs,
                            unsigned short* __restrict__ qin) {
  int q = blockIdx.x, c = threadIdx.x;
  int i = idxs[q];
  float cx = coor[i * 3 + 0] * 0.4f;
  float cy = coor[i * 3 + 1] * 0.4f;
  float cz = coor[i * 3 + 2] * 0.4f;
  float v = feat[(size_t)i * CDIM + c] + cx * pos_w[c] + cy * pos_w[CDIM + c] +
            cz * pos_w[2 * CDIM + c] + pos_b[c];
  qin[q * CDIM + c] = f2bf(v);
}

// ---------------------------------------------------------------------------
// Small WMMA GEMM: C_f32[M][N] = A_bf16[M][K] @ Bt_bf16[N][K] + bias
// (B supplied transposed). grid (M/16, N/16), one wave per 16x16 tile.
// ---------------------------------------------------------------------------
__global__ __launch_bounds__(32) void k_gemm16(
    const unsigned short* __restrict__ A, const unsigned short* __restrict__ Bt,
    const float* __restrict__ bias, float* __restrict__ C, int Kdim, int Ndim) {
  int lane = threadIdx.x & 31, hf = lane >> 4, ml = lane & 15;
  int m0 = blockIdx.x * 16, n0 = blockIdx.y * 16;
  v8f acc = {};
  for (int kb = 0; kb < Kdim / 32; ++kb) {
    FragB a, b;
    loadA(a, A + (m0 + ml) * Kdim, kb * 32, hf);
    loadB(b, Bt + (n0 + ml) * Kdim, kb * 32, hf);
    acc = wmma_bf16(a, b, acc);
  }
  float bv = bias[n0 + ml];
#pragma unroll
  for (int i = 0; i < 8; ++i)
    C[(m0 + i + 8 * hf) * Ndim + n0 + ml] = acc[i] + bv;
}

// Q f32 [256][256] (+bq already) -> Qb bf16 [H][256][64]
__global__ void k_cvt_q(const float* __restrict__ Qt,
                        unsigned short* __restrict__ Qb) {
  int t = blockIdx.x * blockDim.x + threadIdx.x;
  if (t >= TOPKN * CDIM) return;
  int q = t >> 8, hd = t & 255, h = hd >> 6, d = hd & 63;
  Qb[(h * TOPKN + q) * DH + d] = f2bf(Qt[t]);
}

// ---------------------------------------------------------------------------
// Flash-attention partial pass.
// grid (CHUNKS, H, 4 qtiles of 64), block 128 (4 waves x 16 query rows).
// K [H][NPAD][D], V^T [H][D][NPAD]; tiles staged via async global->LDS DMA.
// ---------------------------------------------------------------------------
__global__ __launch_bounds__(128) void k_flash(
    const unsigned short* __restrict__ Qb, const unsigned short* __restrict__ Kn,
    const unsigned short* __restrict__ Vt, float* __restrict__ Opart,
    float* __restrict__ mpart, float* __restrict__ lpart) {
  __shared__ unsigned short sK[KBLK * KSTR];      // [128 keys][64 d]
  __shared__ unsigned short sVt[64 * VTSTR];      // [64 d][128 keys]
  __shared__ unsigned short sP[4 * 16 * PSTR];    // per-wave P [16 q][128 keys]
  const int chunk = blockIdx.x, h = blockIdx.y, qt = blockIdx.z;
  const int tid = threadIdx.x, lane = tid & 31, wvid = tid >> 5;
  const int hf = lane >> 4, ml = lane & 15;
  const int qrow = qt * 64 + wvid * 16 + ml;
  const float scale = 0.125f;                     // 1/sqrt(64)

  FragB aq[2];
#pragma unroll
  for (int kb = 0; kb < 2; ++kb)
    loadA(aq[kb], Qb + (h * TOPKN + qrow) * DH, kb * 32, hf);

  float rmax[8], rsum[8];
  v8f oacc[4];
#pragma unroll
  for (int i = 0; i < 8; ++i) { rmax[i] = -__builtin_inff(); rsum[i] = 0.f; }
#pragma unroll
  for (int d = 0; d < 4; ++d) { v8f z = {}; oacc[d] = z; }

  int blocks = NBLKS - chunk * CHUNKS;
  if (blocks > CHUNKS) blocks = CHUNKS;

  for (int blk = 0; blk < blocks; ++blk) {
    int keyBase = (chunk * CHUNKS + blk) * KBLK;
    __syncthreads();
    { // async DMA: K rows + V^T rows straight into LDS; prefetch next block
      const unsigned short* gK = Kn + ((size_t)h * NPAD + keyBase + tid) * DH;
#pragma unroll
      for (int q = 0; q < 8; ++q)
        async_g2l_b128(sK + tid * KSTR + q * 8, gK + q * 8);
      int r = tid >> 1, off = (tid & 1) * 64;
      const unsigned short* gV = Vt + (size_t)(h * DH + r) * NPAD + keyBase + off;
#pragma unroll
      for (int q = 0; q < 8; ++q)
        async_g2l_b128(sVt + r * VTSTR + off + q * 8, gV + q * 8);
      if (blk + 1 < blocks) {
        __builtin_prefetch(gK + KBLK * DH, 0, 0);
        __builtin_prefetch(gV + KBLK, 0, 0);
      }
      wait_async0();
    }
    __syncthreads();

    // S = Q K^T for 8 key tiles (B-frag = contiguous K row segment)
    v8f st[8];
    for (int kt = 0; kt < 8; ++kt) {
      v8f s = {};
#pragma unroll
      for (int kb = 0; kb < 2; ++kb) {
        FragB bk_;
        loadB(bk_, sK + (kt * 16 + ml) * KSTR, kb * 32, hf);
        s = wmma_bf16(aq[kb], bk_, s);
      }
      int key = keyBase + kt * 16 + ml;
      bool oob = key >= NPTS;
#pragma unroll
      for (int i = 0; i < 8; ++i) st[kt][i] = oob ? -1e30f : s[i] * scale;
    }

    // online softmax update
    float tm[8];
#pragma unroll
    for (int i = 0; i < 8; ++i) {
      float m = st[0][i];
      for (int kt = 1; kt < 8; ++kt) m = fmaxf(m, st[kt][i]);
      m = fmaxf(m, __shfl_xor(m, 1, 32));
      m = fmaxf(m, __shfl_xor(m, 2, 32));
      m = fmaxf(m, __shfl_xor(m, 4, 32));
      m = fmaxf(m, __shfl_xor(m, 8, 32));
      tm[i] = m;
    }
#pragma unroll
    for (int i = 0; i < 8; ++i) {
      float nm = fmaxf(rmax[i], tm[i]);
      float al = __expf(rmax[i] - nm);
      rsum[i] *= al;
      for (int d = 0; d < 4; ++d) oacc[d][i] *= al;
      rmax[i] = nm;
    }
    float ps[8];
#pragma unroll
    for (int i = 0; i < 8; ++i) ps[i] = 0.f;
    for (int kt = 0; kt < 8; ++kt)
#pragma unroll
      for (int i = 0; i < 8; ++i) {
        float p = __expf(st[kt][i] - rmax[i]);
        ps[i] += p;
        sP[wvid * (16 * PSTR) + (i + 8 * hf) * PSTR + kt * 16 + ml] = f2bf(p);
      }
#pragma unroll
    for (int i = 0; i < 8; ++i) {
      float v = ps[i];
      v += __shfl_xor(v, 1, 32);
      v += __shfl_xor(v, 2, 32);
      v += __shfl_xor(v, 4, 32);
      v += __shfl_xor(v, 8, 32);
      rsum[i] += v;
    }

    // O += P V   (A-frag from per-wave P rows, B-frag from V^T rows)
    for (int kk = 0; kk < 4; ++kk) {
      FragB ap;
      loadA(ap, sP + wvid * (16 * PSTR) + ml * PSTR, kk * 32, hf);
      for (int dt = 0; dt < 4; ++dt) {
        FragB bv_;
        loadB(bv_, sVt + (dt * 16 + ml) * VTSTR, kk * 32, hf);
        oacc[dt] = wmma_bf16(ap, bv_, oacc[dt]);
      }
    }
  }

  int pi = (h * 16 + qt * 4 + wvid) * CHUNKS + chunk;
#pragma unroll
  for (int dt = 0; dt < 4; ++dt)
#pragma unroll
    for (int i = 0; i < 8; ++i)
      Opart[(size_t)pi * 1024 + (i + 8 * hf) * 64 + dt * 16 + ml] = oacc[dt][i];
#pragma unroll
  for (int i = 0; i < 8; ++i)
    if (ml == i) {
      mpart[pi * 16 + i + 8 * hf] = rmax[i];
      lpart[pi * 16 + i + 8 * hf] = rsum[i];
    }
}

// ---------------------------------------------------------------------------
// Combine partial softmax chunks -> Obuf f32 [256][256] (q, h*64+d)
// ---------------------------------------------------------------------------
__global__ void k_combine(const float* __restrict__ Opart,
                          const float* __restrict__ mpart,
                          const float* __restrict__ lpart,
                          float* __restrict__ Obuf) {
  int b = blockIdx.x;              // h*16 + qtile16
  int h = b >> 4, qt16 = b & 15;
  int d = threadIdx.x;             // 64
  for (int r = 0; r < 16; ++r) {
    float M = -__builtin_inff();
    for (int c = 0; c < CHUNKS; ++c) M = fmaxf(M, mpart[(b * CHUNKS + c) * 16 + r]);
    float L = 0.f, o = 0.f;
    for (int c = 0; c < CHUNKS; ++c) {
      float w = __expf(mpart[(b * CHUNKS + c) * 16 + r] - M);
      L += lpart[(b * CHUNKS + c) * 16 + r] * w;
      o += Opart[(size_t)(b * CHUNKS + c) * 1024 + r * 64 + d] * w;
    }
    int q = qt16 * 16 + r;
    Obuf[q * CDIM + h * DH + d] = o / L;
  }
}

// replicate the single computed iteration HISTORY_LEN(=4) times
__global__ void k_replicate(const float* __restrict__ src, float* __restrict__ out) {
  int t = blockIdx.x * blockDim.x + threadIdx.x;
  if (t < TOPKN * CDIM) {
    float v = src[t];
    for (int rep = 0; rep < 4; ++rep) out[rep * TOPKN * CDIM + t] = v;
  }
}

// ---------------------------------------------------------------------------
extern "C" void kernel_launch(void* const* d_in, const int* in_sizes, int n_in,
                              void* d_out, int out_size, void* d_ws, size_t ws_size,
                              hipStream_t stream) {
  (void)in_sizes; (void)n_in; (void)out_size; (void)ws_size;
  const int*   coor   = (const int*)d_in[0];
  const float* feat   = (const float*)d_in[1];
  const float* obj_w1 = (const float*)d_in[2];
  const float* obj_b1 = (const float*)d_in[3];
  const float* obj_w2 = (const float*)d_in[4];
  const float* obj_b2 = (const float*)d_in[5];
  const float* pos_w  = (const float*)d_in[6];
  const float* pos_b  = (const float*)d_in[7];
  const float* wq     = (const float*)d_in[8];
  const float* bq     = (const float*)d_in[9];
  const float* wk     = (const float*)d_in[10];
  const float* bk     = (const float*)d_in[11];
  const float* wv     = (const float*)d_in[12];
  const float* bv     = (const float*)d_in[13];
  const float* wo     = (const float*)d_in[14];
  const float* bo     = (const float*)d_in[15];

  char* ws = (char*)d_ws;
  size_t off = 0;
  auto alloc = [&](size_t bytes) -> char* {
    off = (off + 255) & ~(size_t)255;
    char* p = ws + off;
    off += bytes;
    return p;
  };
  unsigned short* Kn    = (unsigned short*)alloc((size_t)HEADS * NPAD * DH * 2);
  unsigned short* Vt    = (unsigned short*)alloc((size_t)HEADS * DH * NPAD * 2);
  float*          obj   = (float*)alloc((size_t)NPTS * 4);
  unsigned short* w1t   = (unsigned short*)alloc(65536 * 2);
  unsigned short* wkt   = (unsigned short*)alloc(65536 * 2);
  unsigned short* wvt   = (unsigned short*)alloc(65536 * 2);
  unsigned short* wqt   = (unsigned short*)alloc(65536 * 2);
  unsigned short* wot   = (unsigned short*)alloc(65536 * 2);
  unsigned*       st    = (unsigned*)alloc(4 * 4);
  unsigned*       hist  = (unsigned*)alloc(4 * 256 * 4);
  unsigned*       Akey  = (unsigned*)alloc(TOPKN * 4);
  int*            Aidx  = (int*)alloc(TOPKN * 4);
  int*            Bidx  = (int*)alloc(512 * 4);
  int*            idxs  = (int*)alloc(TOPKN * 4);
  unsigned short* qin   = (unsigned short*)alloc(65536 * 2);
  float*          Qtmp  = (float*)alloc(65536 * 4);
  unsigned short* Qb    = (unsigned short*)alloc(65536 * 2);
  float*          Opart = (float*)alloc((size_t)HEADS * 16 * CHUNKS * 1024 * 4);
  float*          mpart = (float*)alloc((size_t)HEADS * 16 * CHUNKS * 16 * 4);
  float*          lpart = (float*)alloc((size_t)HEADS * 16 * CHUNKS * 16 * 4);
  float*          Obuf  = (float*)alloc(65536 * 4);
  unsigned short* Obf   = (unsigned short*)alloc(65536 * 2);
  float*          outb  = (float*)alloc(65536 * 4);

  // 1) weights -> bf16, transposed to [out][in] (B-operand layout)
  k_cvt_bf16_t<<<256, 256, 0, stream>>>(obj_w1, w1t);
  k_cvt_bf16_t<<<256, 256, 0, stream>>>(wk, wkt);
  k_cvt_bf16_t<<<256, 256, 0, stream>>>(wv, wvt);
  k_cvt_bf16_t<<<256, 256, 0, stream>>>(wq, wqt);
  k_cvt_bf16_t<<<256, 256, 0, stream>>>(wo, wot);

  // 2) fused objectness + K/V projection (dynamic LDS: feat + weight tiles)
  size_t smem_nkv = (size_t)(128 * FSTR + 256 * WSTR) * 2;
  k_fused_nkv<<<NBLKS, 256, smem_nkv, stream>>>(feat, w1t, wkt, wvt, obj_b1, bk,
                                                bv, obj_w2, obj_b2, obj, Kn, Vt);

  // 3) exact top-256 (radix select + bitonic order)
  k_radix_init<<<1, 256, 0, stream>>>(st, hist);
  for (int p = 0; p < 4; ++p) {
    k_radix_hist<<<256, 256, 0, stream>>>(obj, st, hist, p);
    k_radix_scan<<<1, 32, 0, stream>>>(st, hist, p);
  }
  k_compact<<<256, 256, 0, stream>>>(obj, st, Akey, Aidx, Bidx);
  k_finalize<<<1, 256, 0, stream>>>(st, Akey, Aidx, Bidx, idxs);

  // 4) Q path
  k_build_qin<<<TOPKN, CDIM, 0, stream>>>(coor, feat, pos_w, pos_b, idxs, qin);
  k_gemm16<<<dim3(16, 16), 32, 0, stream>>>(qin, wqt, bq, Qtmp, CDIM, CDIM);
  k_cvt_q<<<256, 256, 0, stream>>>(Qtmp, Qb);

  // 5) flash attention partials + combine
  k_flash<<<dim3(CHUNKS, HEADS, 4), 128, 0, stream>>>(Qb, Kn, Vt, Opart, mpart, lpart);
  k_combine<<<64, 64, 0, stream>>>(Opart, mpart, lpart, Obuf);

  // 6) output projection + replicate HISTORY_LEN copies
  k_cvt_bf16<<<256, 256, 0, stream>>>(Obuf, Obf, 65536);
  k_gemm16<<<dim3(16, 16), 32, 0, stream>>>(Obf, wot, bo, outb, CDIM, CDIM);
  k_replicate<<<256, 256, 0, stream>>>(outb, (float*)d_out);
}
